// RBF_Cell_torch_58875411694167
// MI455X (gfx1250) — matrix operand order repaired and checked by reading, not compiled, not run on hardware
//
#include <hip/hip_runtime.h>

// RBF layer: y[n] = sum_h w[h]*exp(-(|x_n|^2+|c_h|^2-2 x_n.c_h)/(2 sigma_h^2)) + w[H]
// N=131072, F=64, H=512.  Compute-bound (HBM time ~1.5us) -> WMMA bf16 split path.

#define N_ROWS   131072
#define F_DIM    64
#define H_DIM    512
#define PHASE_H  128                       // centers staged in LDS per phase
#define NUM_PHASES (H_DIM / PHASE_H)       // 4
#define TILES_PER_PHASE (PHASE_H / 16)     // 8
#define BLOCK_THREADS 256                  // 8 waves (wave32)
#define WAVES_PER_BLOCK (BLOCK_THREADS / 32)
#define ROWS_PER_BLOCK (WAVES_PER_BLOCK * 16)   // 128

typedef __attribute__((ext_vector_type(16))) __bf16 v16bf;
typedef __attribute__((ext_vector_type(8)))  __bf16 v8bf;
typedef __attribute__((ext_vector_type(8)))  float  v8f;

__device__ __forceinline__ v16bf cat8(v8bf a, v8bf b) {
    return __builtin_shufflevector(a, b, 0,1,2,3,4,5,6,7,8,9,10,11,12,13,14,15);
}

__global__ __launch_bounds__(BLOCK_THREADS)
void rbf_wmma_kernel(const float* __restrict__ x,
                     const float* __restrict__ w,
                     const float* __restrict__ c,
                     const float* __restrict__ sigma,
                     float* __restrict__ y)
{
    // Staged center data: hi/lo bf16 planes + per-center scalars. ~34 KB static LDS.
    __shared__ __bf16 sChi[PHASE_H * F_DIM];
    __shared__ __bf16 sClo[PHASE_H * F_DIM];
    __shared__ float  sCsq[PHASE_H];
    __shared__ float  sCoef[PHASE_H];   // 1/(2*sigma^2)
    __shared__ float  sW[PHASE_H];

    const int tid   = threadIdx.x;
    const int lane  = tid & 31;
    const int wave  = tid >> 5;
    const int g     = lane >> 4;        // half-wave group (0/1)
    const int lrow  = lane & 15;
    const int Mbase = blockIdx.x * ROWS_PER_BLOCK + wave * 16;

    // ---------------- A fragments (per-wave, reused for all 512 centers) -------------
    // 16-bit A 16x32 layout: lane group g: elems 0..7 -> K = kk*32 + g*8 + i,
    //                                     elems 8..15 -> K = kk*32 + 16 + g*8 + (i-8)
    const float* xrow = x + (size_t)(Mbase + lrow) * F_DIM;
    v16bf ahi[2], alo[2];
    float xsq_part = 0.0f;
    #pragma unroll
    for (int kk = 0; kk < 2; ++kk) {
        const int k0 = kk * 32 + g * 8;
        const float4 p0 = *(const float4*)(xrow + k0);
        const float4 p1 = *(const float4*)(xrow + k0 + 4);
        const float4 p2 = *(const float4*)(xrow + k0 + 16);
        const float4 p3 = *(const float4*)(xrow + k0 + 20);
        const float e[16] = { p0.x,p0.y,p0.z,p0.w, p1.x,p1.y,p1.z,p1.w,
                              p2.x,p2.y,p2.z,p2.w, p3.x,p3.y,p3.z,p3.w };
        #pragma unroll
        for (int i = 0; i < 16; ++i) {
            const __bf16 h = (__bf16)e[i];
            ahi[kk][i] = h;
            alo[kk][i] = (__bf16)(e[i] - (float)h);   // residual: x = hi + lo
            xsq_part += e[i] * e[i];                  // exact f32 |x|^2
        }
    }
    // lane L holds half of row (L&15)'s k range; combine halves, then broadcast
    // the 8 row values this lane needs for the C-matrix layout (M = v + 8*g).
    const float xsq_full = xsq_part + __shfl_xor(xsq_part, 16, 32);
    float xsq_row[8];
    #pragma unroll
    for (int v = 0; v < 8; ++v)
        xsq_row[v] = __shfl(xsq_full, v + 8 * g, 32);

    float ysum[8] = {0.f,0.f,0.f,0.f,0.f,0.f,0.f,0.f};

    for (int p = 0; p < NUM_PHASES; ++p) {
        __syncthreads();   // previous phase fully consumed before overwriting LDS
        // ---------------- stage 128 centers into LDS (hi/lo planes + scalars) -------
        if (tid < PHASE_H) {
            const int h = p * PHASE_H + tid;
            const float* crow = c + (size_t)h * F_DIM;
            float csq = 0.0f;
            #pragma unroll
            for (int j8 = 0; j8 < F_DIM / 8; ++j8) {
                const float4 q0 = *(const float4*)(crow + j8 * 8);
                const float4 q1 = *(const float4*)(crow + j8 * 8 + 4);
                const float e[8] = { q0.x,q0.y,q0.z,q0.w, q1.x,q1.y,q1.z,q1.w };
                v8bf hi8, lo8;
                #pragma unroll
                for (int i = 0; i < 8; ++i) {
                    const __bf16 hh = (__bf16)e[i];
                    hi8[i] = hh;
                    lo8[i] = (__bf16)(e[i] - (float)hh);
                    csq += e[i] * e[i];
                }
                *(v8bf*)&sChi[tid * F_DIM + j8 * 8] = hi8;
                *(v8bf*)&sClo[tid * F_DIM + j8 * 8] = lo8;
            }
            sCsq[tid] = csq;
            const float s = sigma[h];
            sCoef[tid] = 0.5f / (s * s);
            sW[tid] = w[h];
        }
        __syncthreads();

        // ---------------- 8 center tiles of 16 per phase -----------------------------
        for (int t = 0; t < TILES_PER_PHASE; ++t) {
            const int cidx = t * 16 + lrow;           // this lane's center (LDS-local)
            const __bf16* bh = &sChi[cidx * F_DIM];
            const __bf16* bl = &sClo[cidx * F_DIM];
            // 16-bit B 32x16 layout (matches A): group g: elems 0..7 -> K=g*8+i, elems 8..15 -> K=16+g*8+(i-8).
            const int off = g * 8;
            const v16bf bhi0 = cat8(*(const v8bf*)(bh + off),      *(const v8bf*)(bh + off + 16));
            const v16bf blo0 = cat8(*(const v8bf*)(bl + off),      *(const v8bf*)(bl + off + 16));
            const v16bf bhi1 = cat8(*(const v8bf*)(bh + 32 + off), *(const v8bf*)(bh + 32 + off + 16));
            const v16bf blo1 = cat8(*(const v8bf*)(bl + 32 + off), *(const v8bf*)(bl + 32 + off + 16));

            // dot = hi.hi + hi.lo + lo.hi  (bf16x2 split, ~f32 accuracy)
            v8f acc0 = {}; v8f acc1 = {};
            acc0 = __builtin_amdgcn_wmma_f32_16x16x32_bf16(false, ahi[0], false, bhi0, (short)0, acc0, false, false);
            acc1 = __builtin_amdgcn_wmma_f32_16x16x32_bf16(false, alo[0], false, bhi0, (short)0, acc1, false, false);
            acc1 = __builtin_amdgcn_wmma_f32_16x16x32_bf16(false, ahi[0], false, blo0, (short)0, acc1, false, false);
            acc0 = __builtin_amdgcn_wmma_f32_16x16x32_bf16(false, ahi[1], false, bhi1, (short)0, acc0, false, false);
            acc1 = __builtin_amdgcn_wmma_f32_16x16x32_bf16(false, alo[1], false, bhi1, (short)0, acc1, false, false);
            acc1 = __builtin_amdgcn_wmma_f32_16x16x32_bf16(false, ahi[1], false, blo1, (short)0, acc1, false, false);

            const float csq  = sCsq[cidx];
            const float coef = sCoef[cidx];
            const float wh   = sW[cidx];
            // C layout: lane L, VGPR v -> row M = v + 8*(L>>4), col N = L&15.
            #pragma unroll
            for (int v = 0; v < 8; ++v) {
                const float dot = acc0[v] + acc1[v];
                const float sq  = xsq_row[v] + csq - 2.0f * dot;
                ysum[v] = __builtin_fmaf(wh, __expf(-sq * coef), ysum[v]);
            }
        }
    }

    // ---------------- reduce over the 16 column-lanes of each half-wave -------------
    #pragma unroll
    for (int v = 0; v < 8; ++v) {
        float s = ysum[v];
        s += __shfl_xor(s, 1, 32);
        s += __shfl_xor(s, 2, 32);
        s += __shfl_xor(s, 4, 32);
        s += __shfl_xor(s, 8, 32);
        ysum[v] = s;
    }
    if (lrow == 0) {
        const float bias = w[H_DIM];                  // w[-1]
        #pragma unroll
        for (int v = 0; v < 8; ++v)
            y[Mbase + v + 8 * g] = ysum[v] + bias;    // lane 0: rows 0..7, lane16: 8..15
    }
}

extern "C" void kernel_launch(void* const* d_in, const int* in_sizes, int n_in,
                              void* d_out, int out_size, void* d_ws, size_t ws_size,
                              hipStream_t stream) {
    (void)in_sizes; (void)n_in; (void)out_size; (void)d_ws; (void)ws_size;
    const float* x  = (const float*)d_in[0];   // [N, F]
    const float* w  = (const float*)d_in[1];   // [H+1, 1]
    const float* c  = (const float*)d_in[2];   // [H, F]
    const float* sg = (const float*)d_in[3];   // [H, 1]
    float* y = (float*)d_out;                  // [N, 1]

    dim3 grid(N_ROWS / ROWS_PER_BLOCK);        // 1024 blocks x 256 threads (8 waves)
    rbf_wmma_kernel<<<grid, BLOCK_THREADS, 0, stream>>>(x, w, c, sg, y);
}
